// _QuantumGate_65481071396941
// MI455X (gfx1250) — matrix operand order repaired and checked by reading, not compiled, not run on hardware
//
#include <hip/hip_runtime.h>

typedef __attribute__((ext_vector_type(16))) _Float16 v16h;
typedef __attribute__((ext_vector_type(8)))  float    v8f;
typedef __attribute__((ext_vector_type(4)))  float    v4f;

#define N_QUBITS 12
#define NANG     36            // 3 * N_QUBITS output columns
#define FDIM     512           // K
#define BATCH    8192          // M
#define MT       16            // rows per wave tile
#define NTILES   (BATCH / MT)  // 512 M-tiles
#define BLOCK    128           // 4 wave32s per block -> spread across more WGPs
#define WPB      (BLOCK / 32)

// angles = x @ W^T + b ; out[b,q] = prod_{j<=q} cos(ang[b,3j]) * cos(ang[b,3j+1])
__global__ __launch_bounds__(BLOCK)
void qgate_wmma_kernel(const float* __restrict__ x,
                       const float* __restrict__ W,
                       const float* __restrict__ bias,
                       float* __restrict__ out)
{
    // W staged once per workgroup, f32 -> f16, row-major (n,k): this is exactly
    // the contiguous-K order the B fragments consume.  36 KB (L2-resident source).
    __shared__ __align__(32) _Float16 sW[NANG * FDIM];
    // zero B-fragment source for padded columns 36..47 of N-tile 2
    __shared__ __align__(32) _Float16 sZero[16];
    // per-wave 16x36 f32 staging of the C tiles for the cross-lane epilogue
    __shared__ float sAng[WPB][MT * NANG];

    const int tid  = threadIdx.x;
    const int lane = tid & 31;     // wave32 (gfx1250)
    const int wid  = tid >> 5;

    for (int idx = tid; idx < NANG * FDIM; idx += BLOCK)
        sW[idx] = (_Float16)W[idx];
    if (tid < 16) sZero[tid] = (_Float16)0.f;
    __syncthreads();

    const int hi   = lane >> 4;    // lane-half select per WMMA VGPR layouts
    const int lmod = lane & 15;

    const int waveGlobal = blockIdx.x * WPB + wid;
    const int nWaves     = gridDim.x * WPB;

    for (int tile = waveGlobal; tile < NTILES; tile += nWaves) {
        const int rowBase = tile * MT;
        v8f c0 = {}, c1 = {}, c2 = {};

        const float* xr = x + (size_t)(rowBase + lmod) * FDIM;

        for (int k0 = 0; k0 < FDIM; k0 += 32) {
            // ---- A fragment: 16x32 f16 (contiguous-K order, same as B) ----
            // lanes 0-15 : halfs 0..15 = K0..K0+15
            // lanes 16-31: halfs 0..15 = K0+16..K0+31
            const int ka = k0 + hi * 16;
            const int kb = ka + 8;
            v4f p0 = *(const v4f*)(xr + ka);
            v4f p1 = *(const v4f*)(xr + ka + 4);
            v4f p2 = *(const v4f*)(xr + kb);
            v4f p3 = *(const v4f*)(xr + kb + 4);
            v16h a;
            #pragma unroll
            for (int i = 0; i < 4; ++i) {
                a[i]      = (_Float16)p0[i];
                a[4 + i]  = (_Float16)p1[i];
                a[8 + i]  = (_Float16)p2[i];
                a[12 + i] = (_Float16)p3[i];
            }

            // ---- B fragments: 32x16 f16, col = ntile*16 + lmod ----
            // lanes 0-15 hold K=k0..k0+15 of their column, lanes 16-31 K=k0+16..k0+31
            const int kBb = k0 + hi * 16;
            const _Float16* b0p = &sW[(0 * 16 + lmod) * FDIM + kBb];
            const _Float16* b1p = &sW[(1 * 16 + lmod) * FDIM + kBb];
            const int col2 = 32 + lmod;
            const _Float16* b2p = (col2 < NANG) ? &sW[col2 * FDIM + kBb] : sZero;
            v16h b0 = *(const v16h*)b0p;
            v16h b1 = *(const v16h*)b1p;
            v16h b2 = *(const v16h*)b2p;

            c0 = __builtin_amdgcn_wmma_f32_16x16x32_f16(false, a, false, b0,
                                                        (short)0, c0, false, false);
            c1 = __builtin_amdgcn_wmma_f32_16x16x32_f16(false, a, false, b1,
                                                        (short)0, c1, false, false);
            c2 = __builtin_amdgcn_wmma_f32_16x16x32_f16(false, a, false, b2,
                                                        (short)0, c2, false, false);
        }

        // ---- epilogue: C layout -> LDS (VGPR r: M = 8*hi + r, N = lmod) ----
        float* ang = sAng[wid];
        #pragma unroll
        for (int r = 0; r < 8; ++r) {
            const int m = hi * 8 + r;
            ang[m * NANG + lmod]      = c0[r];
            ang[m * NANG + 16 + lmod] = c1[r];
            if (lmod < 4) ang[m * NANG + 32 + lmod] = c2[r];  // cols 36..47 are pad
        }
        // cross-lane LDS visibility within the wave (compiler can't see the dep)
        asm volatile("s_wait_dscnt 0x0" ::: "memory");

        if (lane < MT) {
            const float* arow = ang + lane * NANG;
            float*       orow = out + (size_t)(rowBase + lane) * N_QUBITS;
            float prod = 1.f;
            #pragma unroll
            for (int q = 0; q < N_QUBITS; ++q) {
                const float tx = arow[3 * q]     + bias[3 * q];
                const float ty = arow[3 * q + 1] + bias[3 * q + 1];
                prod *= __cosf(tx) * __cosf(ty);
                orow[q] = prod;
            }
        }
        // keep next iteration's LDS writes behind this iteration's reads
        asm volatile("s_wait_dscnt 0x0" ::: "memory");
    }
}

extern "C" void kernel_launch(void* const* d_in, const int* in_sizes, int n_in,
                              void* d_out, int out_size, void* d_ws, size_t ws_size,
                              hipStream_t stream) {
    const float* x    = (const float*)d_in[0];  // (8192, 512) f32
    const float* W    = (const float*)d_in[1];  // (36, 512)   f32
    const float* bias = (const float*)d_in[2];  // (36,)       f32
    float*       out  = (float*)d_out;          // (8192, 12)  f32

    (void)in_sizes; (void)n_in; (void)out_size; (void)d_ws; (void)ws_size;

    // 128 blocks x 4 waves = 512 wave32s (one 16-row tile each), spread across
    // up to 128 WGPs so the per-WGP vector-memory path doesn't throttle the
    // 23.3 TB/s HBM stream of x.
    dim3 grid(128), block(BLOCK);
    hipLaunchKernelGGL(qgate_wmma_kernel, grid, block, 0, stream, x, W, bias, out);
}
